// EncodeThoughtModel_83983790506083
// MI455X (gfx1250) — matrix operand order repaired and checked by reading, not compiled, not run on hardware
//
#include <hip/hip_runtime.h>

// ---------------- types & helpers ----------------
typedef float  v8f  __attribute__((ext_vector_type(8)));
typedef __bf16 v16bf __attribute__((ext_vector_type(16)));
typedef __bf16 v8bf  __attribute__((ext_vector_type(8)));
typedef __bf16 v4bf  __attribute__((ext_vector_type(4)));

#define DMODEL 1024
#define NHEAD  16
#define DHEAD  64
#define BATCH  8
#define SEQ    512
#define DIN    512

// native bf16 convert (gfx1250 has HW cvt; RNE like the reference)
__device__ __forceinline__ __bf16 f2bf(float f) { return (__bf16)f; }

// A-fragment (16x32, lane m = row): K = half*16 .. half*16+15 (contiguous)
__device__ __forceinline__ v16bf load_afrag_f32(const float* p, int half) {
    const float4* q = (const float4*)(p + half * 16);
    float4 a0 = q[0], a1 = q[1];   // K bk..bk+7
    float4 a2 = q[2], a3 = q[3];   // K bk+8..bk+15
    v16bf r;
    r[0] = f2bf(a0.x); r[1] = f2bf(a0.y); r[2] = f2bf(a0.z); r[3] = f2bf(a0.w);
    r[4] = f2bf(a1.x); r[5] = f2bf(a1.y); r[6] = f2bf(a1.z); r[7] = f2bf(a1.w);
    r[8]  = f2bf(a2.x); r[9]  = f2bf(a2.y); r[10] = f2bf(a2.z); r[11] = f2bf(a2.w);
    r[12] = f2bf(a3.x); r[13] = f2bf(a3.y); r[14] = f2bf(a3.z); r[15] = f2bf(a3.w);
    return r;
}
__device__ __forceinline__ v16bf load_afrag_bf(const __bf16* p, int half) {
    union { v16bf v; v8bf h[2]; } u;
    u.h[0] = *(const v8bf*)(p + half * 16);
    u.h[1] = *(const v8bf*)(p + half * 16 + 8);
    return u.v;
}
// B-fragment (32x16, lane n = column): K = half*16 .. half*16+15 (contiguous)
__device__ __forceinline__ v16bf load_bfrag_f32(const float* p, int half) {
    const float4* q = (const float4*)(p + half * 16);
    float4 a0 = q[0], a1 = q[1], a2 = q[2], a3 = q[3];
    v16bf r;
    r[0] = f2bf(a0.x); r[1] = f2bf(a0.y); r[2] = f2bf(a0.z); r[3] = f2bf(a0.w);
    r[4] = f2bf(a1.x); r[5] = f2bf(a1.y); r[6] = f2bf(a1.z); r[7] = f2bf(a1.w);
    r[8]  = f2bf(a2.x); r[9]  = f2bf(a2.y); r[10] = f2bf(a2.z); r[11] = f2bf(a2.w);
    r[12] = f2bf(a3.x); r[13] = f2bf(a3.y); r[14] = f2bf(a3.z); r[15] = f2bf(a3.w);
    return r;
}
__device__ __forceinline__ v16bf load_bfrag_bf(const __bf16* p, int half) {
    union { v16bf v; v8bf h[2]; } u;
    u.h[0] = *(const v8bf*)(p + half * 16);
    u.h[1] = *(const v8bf*)(p + half * 16 + 8);
    return u.v;
}
__device__ __forceinline__ v8f wmma_bf16(v16bf a, v16bf b, v8f c) {
    return __builtin_amdgcn_wmma_f32_16x16x32_bf16(false, a, false, b, (short)0, c, false, false);
}

// ---------------- GEMM: C[M,N] = A[M,K] @ W[K,N] + bias ----------------
#define BM 128
#define BN 128
#define BK 32
__global__ __launch_bounds__(256) void gemm_kernel(
    const float* __restrict__ A, const float* __restrict__ W,
    const float* __restrict__ bias, float* __restrict__ C,
    int M, int N, int K)
{
    __shared__ __align__(16) __bf16 As[2][BM][BK];
    __shared__ __align__(16) __bf16 Bs[2][BN][BK];   // W transposed: Bs[.][n][k]

    const int tid  = threadIdx.x;
    const int wave = tid >> 5, lane = tid & 31;
    const int m0 = blockIdx.y * BM, n0 = blockIdx.x * BN;
    const int wm = (wave >> 1) * 32;   // 4 waves along M
    const int wn = (wave & 1) * 64;    // 2 waves along N
    const int m = lane & 15, half = lane >> 4;

    v8f acc[2][4] = {};
    const int nk = K / BK;             // K is a multiple of 32 in this model

    float4 ra[4], rw[4];               // in-flight tile registers

    // issue global loads for tile at k0 (no wait)
    auto issue_loads = [&](int k0) {
#pragma unroll
        for (int it = 0; it < 4; ++it) {
            int idx = (tid + it * 256) * 4;          // A: 0..4092
            int r = idx >> 5, c = idx & 31;
            int gm = m0 + r;
            ra[it] = make_float4(0.f, 0.f, 0.f, 0.f);
            if (gm < M) {
                const float* ap = &A[(size_t)gm * K + k0 + c];
                ra[it] = *(const float4*)ap;
                if (k0 + 2 * BK < K) __builtin_prefetch(ap + 2 * BK, 0, 1);
            }
        }
#pragma unroll
        for (int it = 0; it < 4; ++it) {
            int idx = (tid + it * 256) * 4;          // W: kk in [0,32), nn in [0,128)
            int kk = idx >> 7, nn = idx & 127;
            int gn = n0 + nn;
            rw[it] = make_float4(0.f, 0.f, 0.f, 0.f);
            if (gn < N) {
                const float* wp = &W[(size_t)(k0 + kk) * N + gn];
                rw[it] = *(const float4*)wp;
                if (k0 + 2 * BK < K) __builtin_prefetch(wp + (size_t)2 * BK * N, 0, 1);
            }
        }
    };
    // convert + store in-flight registers to LDS buffer `buf`
    auto store_tile = [&](int buf) {
#pragma unroll
        for (int it = 0; it < 4; ++it) {
            int idx = (tid + it * 256) * 4;
            int r = idx >> 5, c = idx & 31;
            v4bf o; o[0] = f2bf(ra[it].x); o[1] = f2bf(ra[it].y);
                    o[2] = f2bf(ra[it].z); o[3] = f2bf(ra[it].w);
            *(v4bf*)&As[buf][r][c] = o;
        }
#pragma unroll
        for (int it = 0; it < 4; ++it) {
            int idx = (tid + it * 256) * 4;
            int kk = idx >> 7, nn = idx & 127;
            Bs[buf][nn + 0][kk] = f2bf(rw[it].x);
            Bs[buf][nn + 1][kk] = f2bf(rw[it].y);
            Bs[buf][nn + 2][kk] = f2bf(rw[it].z);
            Bs[buf][nn + 3][kk] = f2bf(rw[it].w);
        }
    };

    // software pipeline: tile k+1 loads are in flight while tile k WMMAs run
    issue_loads(0);
    store_tile(0);
    if (nk > 1) issue_loads(BK);
    for (int kt = 0; kt < nk; ++kt) {
        __syncthreads();
        const int cb = kt & 1;
        v16bf af[2], bfr[4];
#pragma unroll
        for (int mi = 0; mi < 2; ++mi) af[mi] = load_afrag_bf(&As[cb][wm + mi * 16 + m][0], half);
#pragma unroll
        for (int ni = 0; ni < 4; ++ni) bfr[ni] = load_bfrag_bf(&Bs[cb][wn + ni * 16 + m][0], half);
#pragma unroll
        for (int mi = 0; mi < 2; ++mi)
#pragma unroll
            for (int ni = 0; ni < 4; ++ni)
                acc[mi][ni] = wmma_bf16(af[mi], bfr[ni], acc[mi][ni]);
        if (kt + 1 < nk) {
            store_tile((kt + 1) & 1);                  // waits on loads issued last iter
            if (kt + 2 < nk) issue_loads((kt + 2) * BK);
        }
    }
#pragma unroll
    for (int mi = 0; mi < 2; ++mi)
#pragma unroll
        for (int ni = 0; ni < 4; ++ni) {
            int col = n0 + wn + ni * 16 + m;
            if (col >= N) continue;
            float bv = bias ? bias[col] : 0.f;
#pragma unroll
            for (int r = 0; r < 8; ++r) {
                int row = m0 + wm + mi * 16 + r + half * 8;
                if (row < M) C[(size_t)row * N + col] = acc[mi][ni][r] + bv;
            }
        }
}

// ---------------- Attention: O = Qh + softmax(Q Kt / 8) V ----------------
// grid: (ceil(Sq/32), NHEAD, BATCH), block 64 (2 waves, one 16-row q-tile each)
__global__ __launch_bounds__(64) void attn_kernel(
    const float* __restrict__ Q, const float* __restrict__ K,
    const float* __restrict__ V, float* __restrict__ O,
    int Sq, int Sk, unsigned long long q_bstride)
{
    __shared__ __align__(16) __bf16 Vt[DHEAD][SEQ];       // 64 KB
    __shared__ __align__(16) float  Ps[2][16][SEQ];       // 64 KB

    const int wave = threadIdx.x >> 5, lane = threadIdx.x & 31;
    const int b = blockIdx.z, h = blockIdx.y;
    const int m = lane & 15, half = lane >> 4;

    // ---- stage V^T (bf16), float4 along head dim ----
    const float* Vb = V + (size_t)b * Sk * DMODEL + h * DHEAD;
    for (int idx = threadIdx.x * 4; idx < DHEAD * Sk; idx += 64 * 4) {
        int d = idx & 63, s = idx >> 6;
        float4 v = *(const float4*)&Vb[(size_t)s * DMODEL + d];
        Vt[d + 0][s] = f2bf(v.x);
        Vt[d + 1][s] = f2bf(v.y);
        Vt[d + 2][s] = f2bf(v.z);
        Vt[d + 3][s] = f2bf(v.w);
    }
    __syncthreads();

    const int row0 = blockIdx.x * 32 + wave * 16;
    const bool active = row0 < Sq;
    const float* Qb = Q + (size_t)b * q_bstride + h * DHEAD;
    const float* Kb = K + (size_t)b * Sk * DMODEL + h * DHEAD;

    // ---- pass 1: scores into LDS ----
    if (active) {
        const float* qrow = Qb + (size_t)(row0 + m) * DMODEL;
        v16bf aq0 = load_afrag_f32(qrow, half);
        v16bf aq1 = load_afrag_f32(qrow + 32, half);
        for (int kc = 0; kc < Sk; kc += 16) {
            const float* krow = Kb + (size_t)(kc + m) * DMODEL;
            v16bf kb0 = load_bfrag_f32(krow, half);
            v16bf kb1 = load_bfrag_f32(krow + 32, half);
            v8f s = {};
            s = wmma_bf16(aq0, kb0, s);
            s = wmma_bf16(aq1, kb1, s);
#pragma unroll
            for (int r = 0; r < 8; ++r)
                Ps[wave][r + half * 8][kc + m] = s[r] * 0.125f;
        }
    }
    __syncthreads();

    // ---- pass 2: softmax, 2 lanes per row, float4 LDS traffic ----
    if (active) {
        float* prow = Ps[wave][m];
        const int hw = Sk >> 1;               // half width (>= 16, multiple of 4)
        const int h0 = half * hw, h1 = h0 + hw;
        float mx = -1e30f;
        for (int k2 = h0; k2 < h1; k2 += 4) {
            float4 v = *(const float4*)&prow[k2];
            mx = fmaxf(mx, fmaxf(fmaxf(v.x, v.y), fmaxf(v.z, v.w)));
        }
        mx = fmaxf(mx, __shfl_xor(mx, 16, 32));
        float sum = 0.f;
        for (int k2 = h0; k2 < h1; k2 += 4) {
            float4 v = *(const float4*)&prow[k2];
            v.x = __expf(v.x - mx); v.y = __expf(v.y - mx);
            v.z = __expf(v.z - mx); v.w = __expf(v.w - mx);
            *(float4*)&prow[k2] = v;
            sum += v.x + v.y + v.z + v.w;
        }
        sum += __shfl_xor(sum, 16, 32);
        float inv = 1.f / sum;
        for (int k2 = h0; k2 < h1; k2 += 4) {
            float4 v = *(const float4*)&prow[k2];
            v.x *= inv; v.y *= inv; v.z *= inv; v.w *= inv;
            *(float4*)&prow[k2] = v;
        }
    }
    __syncthreads();

    // ---- pass 3: P @ V, add Qh, store ----
    if (active) {
        float* Ob = O + ((size_t)b * Sq + row0) * DMODEL + h * DHEAD;
#pragma unroll
        for (int nt = 0; nt < 4; ++nt) {
            const int n0 = nt * 16;
            v8f acc = {};
            for (int kc = 0; kc < Sk; kc += 32) {
                v16bf ap = load_afrag_f32(&Ps[wave][m][kc], half);
                v16bf bv = load_bfrag_bf(&Vt[n0 + m][kc], half);
                acc = wmma_bf16(ap, bv, acc);
            }
#pragma unroll
            for (int r = 0; r < 8; ++r) {
                int mr = r + half * 8;
                float qv = Qb[(size_t)(row0 + mr) * DMODEL + n0 + m];
                Ob[(size_t)mr * DMODEL + n0 + m] = qv + acc[r];
            }
        }
    }
}

// ---------------- LayerNorm (block per row) ----------------
__global__ __launch_bounds__(256) void ln_kernel(
    float* __restrict__ X, const float* __restrict__ g, const float* __restrict__ bta, int D)
{
    __shared__ float red[256];
    __shared__ float s_mean, s_inv;
    float* row = X + (size_t)blockIdx.x * D;
    int tid = threadIdx.x;
    float s = 0.f;
    for (int i = tid; i < D; i += 256) s += row[i];
    red[tid] = s; __syncthreads();
    for (int off = 128; off; off >>= 1) { if (tid < off) red[tid] += red[tid + off]; __syncthreads(); }
    if (tid == 0) s_mean = red[0] / D;
    __syncthreads();
    float mean = s_mean;
    float v = 0.f;
    for (int i = tid; i < D; i += 256) { float d = row[i] - mean; v += d * d; }
    red[tid] = v; __syncthreads();
    for (int off = 128; off; off >>= 1) { if (tid < off) red[tid] += red[tid + off]; __syncthreads(); }
    if (tid == 0) s_inv = rsqrtf(red[0] / D + 1e-5f);
    __syncthreads();
    float inv = s_inv;
    for (int i = tid; i < D; i += 256) row[i] = (row[i] - mean) * inv * g[i] + bta[i];
}

// ---------------- elementwise kernels ----------------
__global__ void relu_add_kernel(float* __restrict__ X, const float* __restrict__ T, size_t n) {
    size_t i = (size_t)blockIdx.x * blockDim.x + threadIdx.x;
    if (i < n) X[i] += fmaxf(T[i], 0.f);
}
__global__ void add_pe_kernel(float* __restrict__ H, size_t n) {   // n = B*S*D
    size_t i = (size_t)blockIdx.x * blockDim.x + threadIdx.x;
    if (i >= n) return;
    int d = (int)(i % DMODEL);
    int s = (int)((i / DMODEL) % SEQ);
    int p = d >> 1;
    float e = __expf((float)(2 * p) * (-9.210340371976184f / (float)DMODEL));
    float ang = (float)s * e;
    H[i] += (d & 1) ? __cosf(ang) : __sinf(ang);
}
__global__ void assemble_dec_kernel(float* __restrict__ XD, const float* __restrict__ TP,
                                    const float* __restrict__ BOSD, const float* __restrict__ POS, size_t n) {
    size_t i = (size_t)blockIdx.x * blockDim.x + threadIdx.x;
    if (i >= n) return;
    int d = (int)(i % DMODEL);
    int s = (int)((i / DMODEL) % SEQ);
    int b = (int)(i / ((size_t)DMODEL * SEQ));
    float base = (s == 0) ? BOSD[d] : TP[((size_t)b * (SEQ - 1) + (s - 1)) * DMODEL + d];
    XD[i] = base + POS[(size_t)s * DMODEL + d];
}
__global__ void copy_kernel(float* __restrict__ dst, const float* __restrict__ src, size_t n) {
    size_t i = (size_t)blockIdx.x * blockDim.x + threadIdx.x;
    if (i < n) dst[i] = src[i];
}

// ---------------- host orchestration ----------------
extern "C" void kernel_launch(void* const* d_in, const int* in_sizes, int n_in,
                              void* d_out, int out_size, void* d_ws, size_t ws_size,
                              hipStream_t stream) {
    (void)in_sizes; (void)n_in; (void)out_size; (void)ws_size;
    const float* x       = (const float*)d_in[0];
    const float* target  = (const float*)d_in[1];
    const float* ipw     = (const float*)d_in[2];
    const float* ipb     = (const float*)d_in[3];
    const float* wq      = (const float*)d_in[4];
    const float* bq      = (const float*)d_in[5];
    const float* wk      = (const float*)d_in[6];
    const float* bk      = (const float*)d_in[7];
    const float* wv      = (const float*)d_in[8];
    const float* bvv     = (const float*)d_in[9];
    const float* wo      = (const float*)d_in[10];
    const float* bo      = (const float*)d_in[11];
    const float* g0      = (const float*)d_in[12];
    const float* b0      = (const float*)d_in[13];
    const float* g1      = (const float*)d_in[14];
    const float* b1      = (const float*)d_in[15];
    const float* indp    = (const float*)d_in[16];
    const float* slots0  = (const float*)d_in[17];
    const float* decpos  = (const float*)d_in[18];
    const float* opw     = (const float*)d_in[19];
    const float* opb     = (const float*)d_in[20];
    const float* epw     = (const float*)d_in[21];
    const float* epb     = (const float*)d_in[22];
    const float* bos     = (const float*)d_in[23];
    float* out = (float*)d_out;

    const int D = DMODEL, Bx = BATCH, S = SEQ;
    const size_t SZ = (size_t)Bx * S * D;           // 4,194,304 floats
    float* ws    = (float*)d_ws;
    float* bufH  = ws;            // h / encoder output
    float* bufQ  = ws + 1 * SZ;
    float* bufK  = ws + 2 * SZ;
    float* bufV  = ws + 3 * SZ;
    float* bufO  = ws + 4 * SZ;   // tp / recon / generic
    float* bufT  = ws + 5 * SZ;   // FF temp
    float* bufHM = ws + 6 * SZ;   // Hm / decoder X2
    float* bufXD = ws + 7 * SZ;   // decoder stream
    float* bufSl = ws + 8 * SZ;   // slots (B*32*D)
    float* bufBos = bufSl + (size_t)Bx * 32 * D;

    auto gemm = [&](const float* A, const float* W, const float* bias, float* C, int M, int N, int K) {
        dim3 g((N + 127) / 128, (M + 127) / 128);
        gemm_kernel<<<g, 256, 0, stream>>>(A, W, bias, C, M, N, K);
    };
    auto mab = [&](int i, const float* Qx, int Sq, bool qbcast, const float* Kx, int Sk, float* outb) {
        size_t oD = (size_t)i * D, oDD = (size_t)i * D * D;
        int Mq = qbcast ? Sq : Bx * Sq;
        gemm(Qx, wq + oDD, bq + oD, bufQ, Mq, D, D);
        gemm(Kx, wk + oDD, bk + oD, bufK, Bx * Sk, D, D);
        gemm(Kx, wv + oDD, bvv + oD, bufV, Bx * Sk, D, D);
        unsigned long long qbs = qbcast ? 0ull : (unsigned long long)Sq * D;
        attn_kernel<<<dim3((Sq + 31) / 32, NHEAD, Bx), 64, 0, stream>>>(bufQ, bufK, bufV, outb, Sq, Sk, qbs);
        ln_kernel<<<Bx * Sq, 256, 0, stream>>>(outb, g0 + oD, b0 + oD, D);
        gemm(outb, wo + oDD, bo + oD, bufT, Bx * Sq, D, D);
        size_t n = (size_t)Bx * Sq * D;
        relu_add_kernel<<<(unsigned)((n + 255) / 256), 256, 0, stream>>>(outb, bufT, n);
        ln_kernel<<<Bx * Sq, 256, 0, stream>>>(outb, g1 + oD, b1 + oD, D);
    };

    // ---- encoder ----
    gemm(x, ipw, ipb, bufH, Bx * S, D, DIN);
    add_pe_kernel<<<(unsigned)((SZ + 255) / 256), 256, 0, stream>>>(bufH, SZ);
    for (int l = 0; l < 4; ++l) {
        mab(2 * l,     indp + (size_t)l * 64 * D, 64, true,  bufH,  S,  bufHM);  // mab0(I, h)
        mab(2 * l + 1, bufH,                      S,  false, bufHM, 64, bufH);   // mab1(h, Hm)
    }
    // ---- slot bottleneck (index 8) ----
    mab(8, slots0, 32, true, bufH, S, bufSl);

    // ---- decoder input ----
    gemm(target, ipw, ipb, bufO, Bx * (S - 1), D, DIN);   // tp
    gemm(bos, ipw, ipb, bufBos, 1, D, DIN);               // projected BOS
    assemble_dec_kernel<<<(unsigned)((SZ + 255) / 256), 256, 0, stream>>>(bufXD, bufO, bufBos, decpos, SZ);

    // ---- decoder (indices 9..16) ----
    for (int j = 0; j < 4; ++j) {
        mab(9 + 2 * j,  bufXD, S, false, bufXD, S,  bufHM);  // self
        mab(10 + 2 * j, bufHM, S, false, bufSl, 32, bufXD);  // cross to slots
    }

    // ---- output projections ----
    gemm(bufXD, opw, opb, bufO, Bx * S, D, D);            // recon
    gemm(bufO, epw, epb, out, Bx * S, DIN, D);            // projected -> d_out[0 : B*S*DIN)

    // ---- append slots ----
    size_t nsl = (size_t)Bx * 32 * D;
    copy_kernel<<<(unsigned)((nsl + 255) / 256), 256, 0, stream>>>(out + (size_t)Bx * S * DIN, bufSl, nsl);
}
